// LlamaAttention_9440338117100
// MI455X (gfx1250) — matrix operand order repaired and checked by reading, not compiled, not run on hardware
//
#include <hip/hip_runtime.h>

typedef _Float16 h16;
typedef __attribute__((ext_vector_type(8)))  h16  h16x8;
typedef __attribute__((ext_vector_type(16))) h16  h16x16;
typedef __attribute__((ext_vector_type(8)))  float f32x8;

#define S_LEN   2048
#define D_MODEL 1024
#define HD      32
#define NQ      32
#define NKV     8
#define KV_DIM  256   // NKV*HD

#define WMMA_F16(A, B, C) \
  __builtin_amdgcn_wmma_f32_16x16x32_f16(false, (A), false, (B), (short)0, (C), false, false)

// ---------------- CDNA5 async copy helpers -------------------------------------
// GLOBAL_LOAD_ASYNC_TO_LDS_B128 (VGLOBAL op 98): per-lane 16B global -> LDS,
// tracked with ASYNCcnt. LDS dest address is a per-lane VGPR (wave-relative
// byte offset). Generic pointers to __shared__ keep the LDS offset in the low
// 32 bits (aperture base occupies [63:32]), so truncation gives the DS address.
__device__ __forceinline__ unsigned lds_off_u32(const void* p) {
  return (unsigned)(unsigned long long)p;
}
__device__ __forceinline__ void async_b128(const h16* gptr, unsigned lds_off) {
  asm volatile("global_load_async_to_lds_b128 %0, %1, off"
               :: "v"(lds_off), "v"(gptr) : "memory");
}
#define ASYNC_WAIT_3() asm volatile("s_wait_asynccnt 3" ::: "memory")
#define ASYNC_WAIT_2() asm volatile("s_wait_asynccnt 2" ::: "memory")
#define ASYNC_WAIT_0() asm volatile("s_wait_asynccnt 0" ::: "memory")

// ---------------- WMMA fragment loaders (wave32, 16-bit layouts, ISA 7.12.2) ---
// A-matrix 16x32 (MxK): lane L -> row L%16; K = (L/16)*16 + {0..15} (contiguous)
__device__ __forceinline__ h16x16 load_a_frag(const h16* A, int lda,
                                              int row0, int lane) {
  const h16* p = A + (size_t)(row0 + (lane & 15)) * lda + ((lane >> 4) << 4);
  h16x8 lo = *(const h16x8*)(p);        // K = kb .. kb+7
  h16x8 hi = *(const h16x8*)(p + 8);    // K = kb+8 .. kb+15
  return __builtin_shufflevector(lo, hi, 0,1,2,3,4,5,6,7,8,9,10,11,12,13,14,15);
}

// B-matrix 32x16 (KxN): lane L -> col L%16; K = (L/16)*16 + {0..15} (contiguous)
__device__ __forceinline__ h16x16 load_b_frag(const h16* B, int ldb,
                                              int col0, int lane) {
  return *(const h16x16*)(B + (size_t)(col0 + (lane & 15)) * ldb + ((lane >> 4) << 4));
}

// ---------------- GEMM: C[M,N] = A[M,K] * B[N,K]^T  (f16 in, f32 acc/out) ------
// 256 threads (8 waves). Macro-tile 128(M) x 64(N); each wave owns a 32x32
// register tile (2x2 WMMAs, fragments reused 2x). K staged in steps of 32
// through double-buffered LDS via async-to-LDS copies (3 b128 per wave per
// buffer: 2 for A, 1 for B).
__global__ __launch_bounds__(256) void wmma_gemm_nt(const h16* __restrict__ A,
                                                    const h16* __restrict__ B,
                                                    float* __restrict__ C,
                                                    int M, int N, int K) {
  __shared__ h16 As[2][128][32];
  __shared__ h16 Bs[2][64][32];
  const int t    = threadIdx.x;
  const int lane = t & 31;
  const int wave = t >> 5;
  const int mw   = (wave & 3) << 5;   // wave M offset within macro tile
  const int nw   = (wave >> 2) << 5;  // wave N offset within macro tile
  const int n0 = blockIdx.x * 64;
  const int m0 = blockIdx.y * 128;

  const int ra = t >> 2;              // 0..63
  const int ka = (t & 3) << 3;        // {0,8,16,24}

  auto issue = [&](int buf, int k0) {
    async_b128(A + (size_t)(m0 + ra) * K + k0 + ka,      lds_off_u32(&As[buf][ra][ka]));
    async_b128(A + (size_t)(m0 + 64 + ra) * K + k0 + ka, lds_off_u32(&As[buf][64 + ra][ka]));
    async_b128(B + (size_t)(n0 + ra) * K + k0 + ka,      lds_off_u32(&Bs[buf][ra][ka]));
  };

  const int ks = K >> 5;              // K/32 steps (K = 1024 here)
  issue(0, 0);
  issue(1, 32);

  f32x8 acc00 = {}, acc01 = {}, acc10 = {}, acc11 = {};
  for (int i = 0; i < ks; ++i) {
    const int cur = i & 1;
    if (i + 1 < ks) { ASYNC_WAIT_3(); } else { ASYNC_WAIT_0(); }
    __syncthreads();
    h16x16 a0 = load_a_frag(&As[cur][0][0], 32, mw,      lane);
    h16x16 a1 = load_a_frag(&As[cur][0][0], 32, mw + 16, lane);
    h16x16 b0 = load_b_frag(&Bs[cur][0][0], 32, nw,      lane);
    h16x16 b1 = load_b_frag(&Bs[cur][0][0], 32, nw + 16, lane);
    acc00 = WMMA_F16(a0, b0, acc00);
    acc01 = WMMA_F16(a0, b1, acc01);
    acc10 = WMMA_F16(a1, b0, acc10);
    acc11 = WMMA_F16(a1, b1, acc11);
    __syncthreads();
    if (i + 2 < ks) issue(cur, (i + 2) << 5);
  }

  // C/D layout: element r -> row = r + 8*(lane/16), col = lane%16
  const int col   = lane & 15;
  const int rbase = (lane >> 4) << 3;
#pragma unroll
  for (int r = 0; r < 8; ++r) {
    const int row = m0 + mw + rbase + r;
    C[(size_t)row        * N + n0 + nw + col]      = acc00[r];
    C[(size_t)row        * N + n0 + nw + 16 + col] = acc01[r];
    C[(size_t)(row + 16) * N + n0 + nw + col]      = acc10[r];
    C[(size_t)(row + 16) * N + n0 + nw + 16 + col] = acc11[r];
  }
}

// ---------------- elementwise f32 -> f16 ---------------------------------------
__global__ void f32_to_f16(const float* __restrict__ x, h16* __restrict__ y, int n) {
  int i = blockIdx.x * blockDim.x + threadIdx.x;
  if (i < n) y[i] = (h16)x[i];
}

// ---------------- RoPE + layout packing ----------------------------------------
// Qf: [B*S, 1024] f32 (token rows, head-major). Qr: [B, NQ, S, HD] f16.
__global__ void rope_pack_q(const float* __restrict__ Qf, h16* __restrict__ Qr) {
  int i = blockIdx.x * blockDim.x + threadIdx.x;   // B*S*NQ*16 = 2^21
  int d = i & 15;
  int h = (i >> 4) & 31;
  int s = (i >> 9) & 2047;
  int b = i >> 20;
  float inv_freq = powf(10000.0f, -(float)d * (1.0f / 16.0f));
  float ang = (float)s * inv_freq;
  float c = cosf(ang), sn = sinf(ang);
  const float* qp = Qf + ((size_t)b * S_LEN + s) * D_MODEL + h * HD;
  float x1 = qp[d], x2 = qp[d + 16];
  h16* op = Qr + (((size_t)(b * NQ + h)) * S_LEN + s) * HD;
  op[d]      = (h16)(x1 * c - x2 * sn);
  op[d + 16] = (h16)(x2 * c + x1 * sn);
}

// Kf: [B*S, 256] f32. Kr: [B, NKV, S, HD] f16 (d contiguous -> B-fragment friendly).
__global__ void rope_pack_k(const float* __restrict__ Kf, h16* __restrict__ Kr) {
  int i = blockIdx.x * blockDim.x + threadIdx.x;   // B*S*NKV*16 = 2^19
  int d = i & 15;
  int h = (i >> 4) & 7;
  int s = (i >> 7) & 2047;
  int b = i >> 18;
  float inv_freq = powf(10000.0f, -(float)d * (1.0f / 16.0f));
  float ang = (float)s * inv_freq;
  float c = cosf(ang), sn = sinf(ang);
  const float* kp = Kf + ((size_t)b * S_LEN + s) * KV_DIM + h * HD;
  float x1 = kp[d], x2 = kp[d + 16];
  h16* op = Kr + (((size_t)(b * NKV + h)) * S_LEN + s) * HD;
  op[d]      = (h16)(x1 * c - x2 * sn);
  op[d + 16] = (h16)(x2 * c + x1 * sn);
}

// Vf: [B*S, 256] f32. Vt: [B, NKV, HD, S] f16 (transposed so PV B-fragments are
// contiguous along keys).
__global__ void pack_v(const float* __restrict__ Vf, h16* __restrict__ Vt) {
  int i = blockIdx.x * blockDim.x + threadIdx.x;   // B*S*NKV*HD = 2^20
  int d = i & 31;
  int h = (i >> 5) & 7;
  int s = (i >> 8) & 2047;
  int b = i >> 19;
  Vt[(((size_t)(b * NKV + h)) * HD + d) * S_LEN + s] =
      (h16)Vf[((size_t)b * S_LEN + s) * KV_DIM + h * HD + d];
}

// ---------------- Streaming (flash) causal GQA attention -----------------------
// Block = 4 waves; each wave owns a 16-row q tile (block covers 64 q rows of one
// (b,h)). K/V 32-key tiles are staged in double-buffered LDS with async copies
// (1 K + 1 V b128 per thread per buffer => 2 ASYNCcnt ops per wave) and shared
// by all 4 waves. Per 32-key block: 2 WMMAs (Q*K^T), online softmax (width-16
// shuffles; C-fragment rows live in 16-lane halves), P relaid via per-wave LDS
// to A-fragment form, 2 WMMAs (P*V) accumulating into the f32 C operand.
__global__ __launch_bounds__(128) void attn_fa(const h16* __restrict__ Qr,
                                               const h16* __restrict__ Kr,
                                               const h16* __restrict__ Vt,
                                               const int* __restrict__ amask,
                                               h16* __restrict__ AO) {
  __shared__ h16 Kl[2][32][32];
  __shared__ h16 Vl[2][32][32];
  __shared__ h16 Pl[4][16][32];
  const int t    = threadIdx.x;
  const int lane = t & 31;
  const int wave = t >> 5;
  const int b    = blockIdx.z;
  const int h    = blockIdx.y;
  const int hkv  = h >> 2;                      // 32 q-heads / 8 kv-heads
  const int q0   = blockIdx.x * 64 + wave * 16;

  const h16* Qh = Qr + ((size_t)(b * NQ  + h  )) * S_LEN * HD;
  const h16* Kh = Kr + ((size_t)(b * NKV + hkv)) * S_LEN * HD;
  const h16* Vh = Vt + ((size_t)(b * NKV + hkv)) * HD * S_LEN;
  const int* am = amask + b * S_LEN;

  const int rk = t >> 2;               // 0..31 (key row for K / d row for V)
  const int kk = (t & 3) << 3;         // {0,8,16,24}
  auto issue = [&](int buf, int j) {
    async_b128(Kh + (size_t)(j + rk) * HD + kk,  lds_off_u32(&Kl[buf][rk][kk]));
    async_b128(Vh + (size_t)rk * S_LEN + j + kk, lds_off_u32(&Vl[buf][rk][kk]));
  };

  const h16x16 qf = load_a_frag(Qh, HD, q0, lane);   // full head dim in one frag

  f32x8 o0 = {}, o1 = {};
  float m[8], l[8];
#pragma unroll
  for (int r = 0; r < 8; ++r) { m[r] = -3.0e38f; l[r] = 0.0f; }

  const float scale = 0.17677669529663687f;    // 1/sqrt(32)
  const int jend = (blockIdx.x + 1) * 64;      // uniform across block; >= 64
  const int qb   = q0 + ((lane >> 4) << 3);    // q-row base for this half-wave

  issue(0, 0);
  issue(1, 32);

  for (int j = 0; j < jend; j += 32) {
    const int cur = (j >> 5) & 1;
    if (j + 32 < jend) { ASYNC_WAIT_2(); } else { ASYNC_WAIT_0(); }
    __syncthreads();

    h16x16 kb0 = load_b_frag(&Kl[cur][0][0], 32, 0,  lane);
    h16x16 kb1 = load_b_frag(&Kl[cur][0][0], 32, 16, lane);
    f32x8 z = {};
    f32x8 s0 = WMMA_F16(qf, kb0, z);
    f32x8 s1 = WMMA_F16(qf, kb1, z);

    const int key0 = j + (lane & 15);
    const int key1 = key0 + 16;
    const bool ok0 = (am[key0] != 0);
    const bool ok1 = (am[key1] != 0);

#pragma unroll
    for (int r = 0; r < 8; ++r) {
      const int q = qb + r;
      float v0 = (ok0 && key0 <= q) ? s0[r] * scale : -1.0e9f;
      float v1 = (ok1 && key1 <= q) ? s1[r] * scale : -1.0e9f;
      float rmax = fmaxf(v0, v1);
      rmax = fmaxf(rmax, __shfl_xor(rmax, 1, 16));
      rmax = fmaxf(rmax, __shfl_xor(rmax, 2, 16));
      rmax = fmaxf(rmax, __shfl_xor(rmax, 4, 16));
      rmax = fmaxf(rmax, __shfl_xor(rmax, 8, 16));
      const float mn    = fmaxf(m[r], rmax);
      const float alpha = __expf(m[r] - mn);
      const float p0 = __expf(v0 - mn);
      const float p1 = __expf(v1 - mn);
      float rs = p0 + p1;
      rs += __shfl_xor(rs, 1, 16);
      rs += __shfl_xor(rs, 2, 16);
      rs += __shfl_xor(rs, 4, 16);
      rs += __shfl_xor(rs, 8, 16);
      l[r] = l[r] * alpha + rs;
      m[r] = mn;
      o0[r] *= alpha;
      o1[r] *= alpha;
      const int qrl = r + ((lane >> 4) << 3);   // local q row 0..15
      Pl[wave][qrl][lane & 15]        = (h16)p0;
      Pl[wave][qrl][16 + (lane & 15)] = (h16)p1;
    }
    __syncthreads();

    const h16* pp = &Pl[wave][lane & 15][(lane >> 4) << 4];
    h16x8 plo = *(const h16x8*)pp;
    h16x8 phi = *(const h16x8*)(pp + 8);
    h16x16 pf = __builtin_shufflevector(plo, phi, 0,1,2,3,4,5,6,7,8,9,10,11,12,13,14,15);

    // V B-fragments from LDS: lane -> d = tile*16 + lane%16, 16 contiguous keys
    h16x16 vb0 = *(const h16x16*)&Vl[cur][lane & 15][(lane >> 4) << 4];
    h16x16 vb1 = *(const h16x16*)&Vl[cur][16 + (lane & 15)][(lane >> 4) << 4];
    o0 = WMMA_F16(pf, vb0, o0);
    o1 = WMMA_F16(pf, vb1, o1);

    __syncthreads();                         // everyone done with Kl/Vl/Pl[cur]
    if (j + 64 < jend) issue(cur, j + 64);
  }

  // Normalize and store as [B, S, NQ*HD] (token-major, head-major cols) in f16.
  const int col = lane & 15;
#pragma unroll
  for (int r = 0; r < 8; ++r) {
    const int tok = qb + r;
    const float inv = 1.0f / l[r];
    size_t base = ((size_t)(b * S_LEN + tok)) * D_MODEL + h * HD;
    AO[base + col]      = (h16)(o0[r] * inv);
    AO[base + 16 + col] = (h16)(o1[r] * inv);
  }
}

// ---------------- host-side orchestration --------------------------------------
extern "C" void kernel_launch(void* const* d_in, const int* in_sizes, int n_in,
                              void* d_out, int out_size, void* d_ws, size_t ws_size,
                              hipStream_t stream) {
  const float* hidden = (const float*)d_in[0];
  const int*   amask  = (const int*)  d_in[1];
  const float* Wq     = (const float*)d_in[2];
  const float* Wk     = (const float*)d_in[3];
  const float* Wv     = (const float*)d_in[4];
  const float* Wo     = (const float*)d_in[5];
  float*       out    = (float*)d_out;

  const int B = 2, S = S_LEN, D = D_MODEL;
  const int TOK = B * S;   // 4096

  char* ws = (char*)d_ws;
  size_t off = 0;
  auto alloc = [&](size_t bytes) {
    void* p = ws + off;
    off = (off + bytes + 255) & ~(size_t)255;
    return p;
  };

  h16*   hbf = (h16*)  alloc((size_t)TOK * D * 2);
  h16*   WqB = (h16*)  alloc((size_t)D * D * 2);
  h16*   WkB = (h16*)  alloc((size_t)KV_DIM * D * 2);
  h16*   WvB = (h16*)  alloc((size_t)KV_DIM * D * 2);
  h16*   WoB = (h16*)  alloc((size_t)D * D * 2);
  float* Qf  = (float*)alloc((size_t)TOK * D * 4);
  float* Kf  = (float*)alloc((size_t)TOK * KV_DIM * 4);
  float* Vf  = (float*)alloc((size_t)TOK * KV_DIM * 4);
  h16*   Qr  = (h16*)  alloc((size_t)TOK * D * 2);
  h16*   Kr  = (h16*)  alloc((size_t)TOK * KV_DIM * 2);
  h16*   Vt  = (h16*)  alloc((size_t)TOK * KV_DIM * 2);
  h16*   AO  = (h16*)  alloc((size_t)TOK * D * 2);

  int n;
  n = TOK * D;    f32_to_f16<<<(n + 255) / 256, 256, 0, stream>>>(hidden, hbf, n);
  n = D * D;      f32_to_f16<<<(n + 255) / 256, 256, 0, stream>>>(Wq, WqB, n);
  n = KV_DIM * D; f32_to_f16<<<(n + 255) / 256, 256, 0, stream>>>(Wk, WkB, n);
  n = KV_DIM * D; f32_to_f16<<<(n + 255) / 256, 256, 0, stream>>>(Wv, WvB, n);
  n = D * D;      f32_to_f16<<<(n + 255) / 256, 256, 0, stream>>>(Wo, WoB, n);

  // QKV projections (A @ W^T): macro-tiles 128(M) x 64(N)
  wmma_gemm_nt<<<dim3(D / 64,      TOK / 128), 256, 0, stream>>>(hbf, WqB, Qf, TOK, D, D);
  wmma_gemm_nt<<<dim3(KV_DIM / 64, TOK / 128), 256, 0, stream>>>(hbf, WkB, Kf, TOK, KV_DIM, D);
  wmma_gemm_nt<<<dim3(KV_DIM / 64, TOK / 128), 256, 0, stream>>>(hbf, WvB, Vf, TOK, KV_DIM, D);

  // RoPE + attention-friendly packing
  n = B * S * NQ * 16;  rope_pack_q<<<n / 256, 256, 0, stream>>>(Qf, Qr);
  n = B * S * NKV * 16; rope_pack_k<<<n / 256, 256, 0, stream>>>(Kf, Kr);
  n = B * S * NKV * HD; pack_v<<<n / 256, 256, 0, stream>>>(Vf, Vt);

  // Streaming causal GQA attention
  attn_fa<<<dim3(S / 64, NQ, B), 128, 0, stream>>>(Qr, Kr, Vt, amask, AO);

  // Output projection
  wmma_gemm_nt<<<dim3(D / 64, TOK / 128), 256, 0, stream>>>(AO, WoB, out, TOK, D, D);
}
